// GatNet_41764261987086
// MI455X (gfx1250) — matrix operand order repaired and checked by reading, not compiled, not run on hardware
//
#include <hip/hip_runtime.h>
#include <hip/hip_bf16.h>
#include <stdint.h>

#define N_NODES 100000
#define N_EDGES 3200000
#define IN_F    128
#define HID_F   64
#define OUT_F   32
#define NEG_SLOPE 0.2f

typedef __attribute__((ext_vector_type(16))) _Float16 v16h;
typedef __attribute__((ext_vector_type(8)))  float    v8f;

// ---------------------------------------------------------------------------
// Monotone uint key for float atomic-max (works for negatives, init key = 0
// is below key(-inf) = 0x007FFFFF, and every node has a self-loop edge).
// ---------------------------------------------------------------------------
__device__ __forceinline__ unsigned fkey(float f) {
  unsigned u = __float_as_uint(f);
  return (u & 0x80000000u) ? ~u : (u | 0x80000000u);
}
__device__ __forceinline__ float fdecode(unsigned k) {
  return (k & 0x80000000u) ? __uint_as_float(k & 0x7FFFFFFFu)
                           : __uint_as_float(~k);
}

__device__ __forceinline__ float lrelu(float x) {
  return x > 0.f ? x : NEG_SLOPE * x;
}

// ---------------------------------------------------------------------------
// C[M x F] = A[M x K] @ B[K x F], f32 in/out, f16 WMMA with f32 accumulate.
// One wave computes one 16x16 tile; K stepped by 32.
// A frag (16x32 f16, ISA 7.12.2): lane M = lane&15, K-base = (lane>>4)*8,
//   elems 0..7 -> K = base+i, elems 8..15 -> K = base+16+i.
// B frag (32x16 f16): lane N = lane&15, K-base = (lane>>4)*8, same K order as A.
// D (16x16 f32): VGPR r -> row = m0 + r + 8*(lane>=16), col = n0 + (lane&15).
// M = 100000 is an exact multiple of 16 -> EXEC all-1s at every WMMA.
// ---------------------------------------------------------------------------
template <int K, int F>
__global__ __launch_bounds__(256) void gemm_wmma_kernel(
    const float* __restrict__ A, const float* __restrict__ B,
    float* __restrict__ C) {
  const int lane  = threadIdx.x & 31;
  const int wave  = threadIdx.x >> 5;
  const int mtile = blockIdx.x * 8 + wave;
  if (mtile * 16 >= N_NODES) return;              // wave-uniform branch
  const int m0 = mtile * 16;
  const int n0 = blockIdx.y * 16;

  const int rowA  = m0 + (lane & 15);
  const int colB  = n0 + (lane & 15);
  const int kaoff = (lane >> 4) * 8;
  const int kboff = (lane >> 4) * 8;

  v8f acc = {};
#pragma unroll
  for (int k0 = 0; k0 < K; k0 += 32) {
    v16h a, b;
    const float* ap = A + (size_t)rowA * K + k0 + kaoff;
#pragma unroll
    for (int i = 0; i < 8; ++i) {
      a[i]     = (_Float16)ap[i];
      a[i + 8] = (_Float16)ap[i + 16];
    }
    const float* bp = B + (size_t)(k0 + kboff) * F + colB;
#pragma unroll
    for (int i = 0; i < 8; ++i) {
      b[i]     = (_Float16)bp[(size_t)i * F];
      b[i + 8] = (_Float16)bp[(size_t)(i + 16) * F];
    }
    acc = __builtin_amdgcn_wmma_f32_16x16x32_f16(
        /*neg_a=*/false, a, /*neg_b=*/false, b,
        /*c_mod=*/(short)0, acc, /*reuse_a=*/false, /*reuse_b=*/false);
  }

  const int row0 = m0 + (lane >> 4) * 8;
  const int col  = n0 + (lane & 15);
#pragma unroll
  for (int r = 0; r < 8; ++r)
    C[(size_t)(row0 + r) * F + col] = acc[r];
}

// ---------------------------------------------------------------------------
// Per-node attention scalars: as[n] = h[n,:].a_src, ad[n] = h[n,:].a_dst.
// One wave per node, shfl-xor tree reduction (wave32).
// ---------------------------------------------------------------------------
template <int F>
__global__ __launch_bounds__(256) void alpha_kernel(
    const float* __restrict__ H, const float* __restrict__ a_src,
    const float* __restrict__ a_dst, float* __restrict__ as,
    float* __restrict__ ad) {
  const int lane = threadIdx.x & 31;
  const int node = (int)((blockIdx.x * (long long)blockDim.x + threadIdx.x) >> 5);
  if (node >= N_NODES) return;
  const float* h = H + (size_t)node * F;
  float ps = 0.f, pd = 0.f;
#pragma unroll
  for (int f = lane; f < F; f += 32) {
    float v = h[f];
    ps += v * a_src[f];
    pd += v * a_dst[f];
  }
#pragma unroll
  for (int m = 16; m >= 1; m >>= 1) {
    ps += __shfl_xor(ps, m, 32);
    pd += __shfl_xor(pd, m, 32);
  }
  if (lane == 0) { as[node] = ps; ad[node] = pd; }
}

// ---------------------------------------------------------------------------
// Pass A: segment max of leaky-relu logits into mkey[dst] (uint key atomics).
// One thread per edge (incl. N self-loops appended after the E real edges).
// ---------------------------------------------------------------------------
__global__ __launch_bounds__(256) void edge_max_kernel(
    const long long* __restrict__ ei, const float* __restrict__ as,
    const float* __restrict__ ad, unsigned* __restrict__ mkey) {
  long long t = blockIdx.x * (long long)blockDim.x + threadIdx.x;
  if (t >= (long long)N_EDGES + N_NODES) return;
  int src, dst;
  if (t < N_EDGES) { src = (int)ei[t]; dst = (int)ei[(long long)N_EDGES + t]; }
  else             { src = dst = (int)(t - N_EDGES); }
  atomicMax(&mkey[dst], fkey(lrelu(as[src] + ad[dst])));
}

// ---------------------------------------------------------------------------
// Pass B: w = exp(logit - max[dst]); ssum[dst] += w; agg[dst,:] += w*h[src,:].
// One wave per edge; F feature atomics spread across 32 lanes.
// agg/h/ssum are L2-resident (<< 192 MB) so atomics resolve in L2.
// ---------------------------------------------------------------------------
template <int F>
__global__ __launch_bounds__(256) void edge_agg_kernel(
    const long long* __restrict__ ei, const float* __restrict__ as,
    const float* __restrict__ ad, const unsigned* __restrict__ mkey,
    const float* __restrict__ H, float* __restrict__ ssum,
    float* __restrict__ agg) {
  const int lane = threadIdx.x & 31;
  long long e = (blockIdx.x * (long long)blockDim.x + threadIdx.x) >> 5;
  if (e >= (long long)N_EDGES + N_NODES) return;
  int src, dst;
  if (e < N_EDGES) { src = (int)ei[e]; dst = (int)ei[(long long)N_EDGES + e]; }
  else             { src = dst = (int)(e - N_EDGES); }
  float lg = lrelu(as[src] + ad[dst]);
  float w  = __expf(lg - fdecode(mkey[dst]));
  if (lane == 0) atomicAdd(&ssum[dst], w);
  const float* h = H + (size_t)src * F;
  float*       o = agg + (size_t)dst * F;
#pragma unroll
  for (int f = lane; f < F; f += 32)
    atomicAdd(&o[f], w * h[f]);
}

// ---------------------------------------------------------------------------
// Layer-1 finalize: out = relu(agg/ssum + b)
// ---------------------------------------------------------------------------
__global__ __launch_bounds__(256) void finalize_relu_kernel(
    const float* __restrict__ agg, const float* __restrict__ ssum,
    const float* __restrict__ b, float* __restrict__ out) {
  long long t = blockIdx.x * (long long)blockDim.x + threadIdx.x;
  if (t >= (long long)N_NODES * HID_F) return;
  int n = (int)(t / HID_F), f = (int)(t % HID_F);
  float v = agg[t] / ssum[n] + b[f];
  out[t] = v > 0.f ? v : 0.f;
}

// ---------------------------------------------------------------------------
// Layer-2 finalize + row softmax over OUT_F=32: one wave per node,
// lane == feature, shfl-xor max & sum.
// ---------------------------------------------------------------------------
__global__ __launch_bounds__(256) void finalize_softmax_kernel(
    const float* __restrict__ agg, const float* __restrict__ ssum,
    const float* __restrict__ b, float* __restrict__ out) {
  const int lane = threadIdx.x & 31;
  const int node = (int)((blockIdx.x * (long long)blockDim.x + threadIdx.x) >> 5);
  if (node >= N_NODES) return;
  float v = agg[(size_t)node * OUT_F + lane] / ssum[node] + b[lane];
  float mx = v;
#pragma unroll
  for (int m = 16; m >= 1; m >>= 1) mx = fmaxf(mx, __shfl_xor(mx, m, 32));
  float e = __expf(v - mx);
  float s = e;
#pragma unroll
  for (int m = 16; m >= 1; m >>= 1) s += __shfl_xor(s, m, 32);
  out[(size_t)node * OUT_F + lane] = e / s;
}

// ---------------------------------------------------------------------------
// Host driver
// ---------------------------------------------------------------------------
extern "C" void kernel_launch(void* const* d_in, const int* in_sizes, int n_in,
                              void* d_out, int out_size, void* d_ws,
                              size_t ws_size, hipStream_t stream) {
  (void)in_sizes; (void)n_in; (void)out_size; (void)ws_size;
  const float*     x     = (const float*)d_in[0];
  const long long* ei    = (const long long*)d_in[1];   // int64 [2, E]
  const float*     W1    = (const float*)d_in[2];
  const float*     asrc1 = (const float*)d_in[3];
  const float*     adst1 = (const float*)d_in[4];
  const float*     b1    = (const float*)d_in[5];
  const float*     W2    = (const float*)d_in[6];
  const float*     asrc2 = (const float*)d_in[7];
  const float*     adst2 = (const float*)d_in[8];
  const float*     b2    = (const float*)d_in[9];
  float*           out   = (float*)d_out;

  // Workspace carve-up (all 256B aligned).
  char*  ws  = (char*)d_ws;
  size_t off = 0;
  auto carve = [&](size_t bytes) {
    char* p = ws + off;
    off += (bytes + 255) & ~(size_t)255;
    return p;
  };
  float*    h1   = (float*)carve((size_t)N_NODES * HID_F * 4);  // x@W1
  float*    x2   = (float*)carve((size_t)N_NODES * HID_F * 4);  // layer-1 output
  float*    agg  = (float*)carve((size_t)N_NODES * HID_F * 4);  // shared agg buf
  float*    h2   = (float*)carve((size_t)N_NODES * OUT_F * 4);  // x2@W2
  float*    as   = (float*)carve((size_t)N_NODES * 4);
  float*    ad   = (float*)carve((size_t)N_NODES * 4);
  float*    ssum = (float*)carve((size_t)N_NODES * 4);
  unsigned* mkey = (unsigned*)carve((size_t)N_NODES * 4);

  const long long EN = (long long)N_EDGES + N_NODES;
  const int  blk        = 256;
  const int  grid_nodeW = (int)((N_NODES + 7) / 8);       // 1 wave/node
  const int  grid_edgeT = (int)((EN + blk - 1) / blk);    // 1 thread/edge
  const int  grid_edgeW = (int)((EN + 7) / 8);            // 1 wave/edge

  // ---------------- Layer 1 ----------------
  {
    dim3 g((N_NODES / 16 + 7) / 8, HID_F / 16);
    gemm_wmma_kernel<IN_F, HID_F><<<g, blk, 0, stream>>>(x, W1, h1);
  }
  alpha_kernel<HID_F><<<grid_nodeW, blk, 0, stream>>>(h1, asrc1, adst1, as, ad);

  hipMemsetAsync(mkey, 0, (size_t)N_NODES * 4, stream);
  hipMemsetAsync(ssum, 0, (size_t)N_NODES * 4, stream);
  hipMemsetAsync(agg,  0, (size_t)N_NODES * HID_F * 4, stream);

  edge_max_kernel<<<grid_edgeT, blk, 0, stream>>>(ei, as, ad, mkey);
  edge_agg_kernel<HID_F><<<grid_edgeW, blk, 0, stream>>>(ei, as, ad, mkey, h1,
                                                         ssum, agg);
  {
    long long tot = (long long)N_NODES * HID_F;
    finalize_relu_kernel<<<(int)((tot + blk - 1) / blk), blk, 0, stream>>>(
        agg, ssum, b1, x2);
  }

  // ---------------- Layer 2 ----------------
  {
    dim3 g((N_NODES / 16 + 7) / 8, OUT_F / 16);
    gemm_wmma_kernel<HID_F, OUT_F><<<g, blk, 0, stream>>>(x2, W2, h2);
  }
  alpha_kernel<OUT_F><<<grid_nodeW, blk, 0, stream>>>(h2, asrc2, adst2, as, ad);

  hipMemsetAsync(mkey, 0, (size_t)N_NODES * 4, stream);
  hipMemsetAsync(ssum, 0, (size_t)N_NODES * 4, stream);
  hipMemsetAsync(agg,  0, (size_t)N_NODES * OUT_F * 4, stream);

  edge_max_kernel<<<grid_edgeT, blk, 0, stream>>>(ei, as, ad, mkey);
  edge_agg_kernel<OUT_F><<<grid_edgeW, blk, 0, stream>>>(ei, as, ad, mkey, h2,
                                                         ssum, agg);
  finalize_softmax_kernel<<<grid_nodeW, blk, 0, stream>>>(agg, ssum, b2, out);
}
